// VectorQuantizer_88235808129600
// MI455X (gfx1250) — matrix operand order repaired and checked by reading, not compiled, not run on hardware
//
#include <hip/hip_runtime.h>
#include <hip/hip_bf16.h>

typedef __attribute__((ext_vector_type(16))) __bf16      bf16x16;
typedef __attribute__((ext_vector_type(8)))  float       f32x8;
typedef __attribute__((ext_vector_type(4)))  unsigned int u32x4;

#define NTOK   16384      // B*T
#define GG     8
#define KK     1024
#define DC     128
#define LROW   136        // padded LDS row (bank-conflict avoidance)
#define NELEM  16777216   // B*T*G*DC
#define IDXOFF 16777218   // quantized + 2 losses

union AFrag { bf16x16 v; u32x4 q[2]; };

__device__ __forceinline__ unsigned short f2bf(float f) {
    unsigned int u = __float_as_uint(f);
    unsigned int r = u + 0x7FFFu + ((u >> 16) & 1u);   // round-to-nearest-even
    return (unsigned short)(r >> 16);
}
__device__ __forceinline__ unsigned pack2bf(float a, float b) {
    return (unsigned)f2bf(a) | ((unsigned)f2bf(b) << 16);
}

// ---- prep: codebook fp32 -> bf16 + row norms c2[g*K+k]; zero loss acc ----
__global__ __launch_bounds__(256) void vq_prep_cb(const float* __restrict__ cb,
                                                  unsigned short* __restrict__ cbb,
                                                  float* __restrict__ c2,
                                                  double* __restrict__ lossAcc) {
    if (blockIdx.x == 0 && threadIdx.x == 0) *lossAcc = 0.0;
    const int row  = blockIdx.x * 8 + (threadIdx.x >> 5);   // [0, G*K)
    const int lane = threadIdx.x & 31;
    const float4 f = *(const float4*)(cb + (size_t)row * DC + lane * 4);
    float s = f.x * f.x + f.y * f.y + f.z * f.z + f.w * f.w;
    #pragma unroll
    for (int m = 1; m < 32; m <<= 1) s += __shfl_xor(s, m, 32);
    if (lane == 0) c2[row] = s;
    ushort4 o;
    o.x = f2bf(f.x); o.y = f2bf(f.y); o.z = f2bf(f.z); o.w = f2bf(f.w);
    *(ushort4*)(cbb + (size_t)row * DC + lane * 4) = o;
}

// ---- main: WMMA scoring + argmin + gather + loss ----
// 512 blocks: 64 blocks/group, each wave owns 32 tokens (2 A-tile sets)
__global__ __launch_bounds__(256) void vq_main(const unsigned short* __restrict__ cbb,
                                               const float* __restrict__ c2g,
                                               const float* __restrict__ xf,
                                               const float* __restrict__ cbf,
                                               float* __restrict__ out,
                                               double* __restrict__ lossAcc) {
    extern __shared__ char smem[];
    unsigned short* cbl = (unsigned short*)smem;              // K x LROW bf16
    float*          c2l = (float*)(smem + KK * LROW * 2);     // K floats

    const int g        = blockIdx.x >> 6;          // 64 blocks per group
    const int tileBase = (blockIdx.x & 63) * 16;   // 16 token-tiles per block
    const int tid      = threadIdx.x;

    // stage this group's codebook (bf16) into LDS via async global->LDS copies
    {
        const unsigned short* src = cbb + (size_t)g * KK * DC;
        for (int chunk = tid; chunk < KK * (DC / 8); chunk += 256) {
            int row = chunk >> 4;
            int c8  = chunk & 15;
            unsigned lds = (unsigned)(row * LROW + c8 * 8) * 2u;   // dyn-LDS base = 0
            unsigned long long ga =
                (unsigned long long)(uintptr_t)(src + (size_t)row * DC + c8 * 8);
            asm volatile("global_load_async_to_lds_b128 %0, %1, off"
                         :: "v"(lds), "v"(ga) : "memory");
        }
        const float* c2src = c2g + g * KK;
        for (int i = tid; i < KK; i += 256) c2l[i] = c2src[i];
        asm volatile("s_wait_asynccnt 0x0" ::: "memory");
    }
    __syncthreads();

    const int wave   = tid >> 5;
    const int lane   = tid & 31;
    const int tt0    = tileBase + wave * 2;        // first of two token tiles
    const int rowIn  = lane & 15;
    const int halfHi = lane >> 4;

    // Two resident A fragment sets: tokens [tt0*16, tt0*16+32)
    // fp32 rows loaded once and packed to bf16 in-register (fused conversion)
    AFrag a0[4], a1[4];
    {
        const int token = tt0 * 16 + rowIn;
        const float* xr0 = xf + ((size_t)token * GG + g) * DC;
        const float* xr1 = xr0 + (size_t)16 * GG * DC;
        const int k0 = halfHi * 8;
        #pragma unroll
        for (int c = 0; c < 4; ++c) {
            float4 fa, fb;
            fa = *(const float4*)(xr0 + c * 32 + k0);
            fb = *(const float4*)(xr0 + c * 32 + k0 + 4);
            a0[c].q[0] = (u32x4){pack2bf(fa.x, fa.y), pack2bf(fa.z, fa.w),
                                 pack2bf(fb.x, fb.y), pack2bf(fb.z, fb.w)};
            fa = *(const float4*)(xr0 + c * 32 + k0 + 16);
            fb = *(const float4*)(xr0 + c * 32 + k0 + 20);
            a0[c].q[1] = (u32x4){pack2bf(fa.x, fa.y), pack2bf(fa.z, fa.w),
                                 pack2bf(fb.x, fb.y), pack2bf(fb.z, fb.w)};
            fa = *(const float4*)(xr1 + c * 32 + k0);
            fb = *(const float4*)(xr1 + c * 32 + k0 + 4);
            a1[c].q[0] = (u32x4){pack2bf(fa.x, fa.y), pack2bf(fa.z, fa.w),
                                 pack2bf(fb.x, fb.y), pack2bf(fb.z, fb.w)};
            fa = *(const float4*)(xr1 + c * 32 + k0 + 16);
            fb = *(const float4*)(xr1 + c * 32 + k0 + 20);
            a1[c].q[1] = (u32x4){pack2bf(fa.x, fa.y), pack2bf(fa.z, fa.w),
                                 pack2bf(fb.x, fb.y), pack2bf(fb.z, fb.w)};
        }
    }

    float best0[8], best1[8]; int bk0[8], bk1[8];
    #pragma unroll
    for (int r = 0; r < 8; ++r) {
        best0[r] = 3.4e38f; best1[r] = 3.4e38f; bk0[r] = 0; bk1[r] = 0;
    }

    const int kb = halfHi * 8;
    for (int nt = 0; nt < 64; ++nt) {
        const int col = nt * 16 + rowIn;
        f32x8 acc0 = {0.f,0.f,0.f,0.f,0.f,0.f,0.f,0.f};
        f32x8 acc1 = {0.f,0.f,0.f,0.f,0.f,0.f,0.f,0.f};
        #pragma unroll
        for (int c = 0; c < 4; ++c) {             // each B fragment feeds 2 WMMAs
            AFrag b;
            const unsigned short* p = cbl + col * LROW + c * 32 + kb;
            b.q[0] = *(const u32x4*)p;
            b.q[1] = *(const u32x4*)(p + 16);
            acc0 = __builtin_amdgcn_wmma_f32_16x16x32_bf16(false, a0[c].v, false, b.v,
                                                           (short)0, acc0, false, false);
            acc1 = __builtin_amdgcn_wmma_f32_16x16x32_bf16(false, a1[c].v, false, b.v,
                                                           (short)0, acc1, false, false);
        }
        const float c2v = c2l[col];
        #pragma unroll
        for (int r = 0; r < 8; ++r) {
            float s0 = c2v - 2.0f * acc0[r];
            float s1 = c2v - 2.0f * acc1[r];
            if (s0 < best0[r]) { best0[r] = s0; bk0[r] = col; }
            if (s1 < best1[r]) { best1[r] = s1; bk1[r] = col; }
        }
    }

    // cross-lane argmin within each 16-lane half, then broadcast
    int idxAll[32];
    #pragma unroll
    for (int r = 0; r < 8; ++r) {
        float v0 = best0[r]; int k0 = bk0[r];
        float v1 = best1[r]; int k1 = bk1[r];
        #pragma unroll
        for (int m = 1; m < 16; m <<= 1) {
            float vo0 = __shfl_xor(v0, m, 32); int ko0 = __shfl_xor(k0, m, 32);
            float vo1 = __shfl_xor(v1, m, 32); int ko1 = __shfl_xor(k1, m, 32);
            if (vo0 < v0 || (vo0 == v0 && ko0 < k0)) { v0 = vo0; k0 = ko0; }
            if (vo1 < v1 || (vo1 == v1 && ko1 < k1)) { v1 = vo1; k1 = ko1; }
        }
        idxAll[r]      = __shfl(k0, 0, 32);
        idxAll[r + 8]  = __shfl(k0, 16, 32);
        idxAll[r + 16] = __shfl(k1, 0, 32);
        idxAll[r + 24] = __shfl(k1, 16, 32);
    }

    if (lane == 0) {
        float* outIdx = out + IDXOFF;
        #pragma unroll
        for (int m = 0; m < 32; ++m)
            outIdx[(size_t)(tt0 * 16 + m) * GG + g] = (float)idxAll[m];
    }

    // exact fp32 gather of q + loss accumulation (32 rows)
    float lsum = 0.0f;
    #pragma unroll
    for (int m = 0; m < 32; ++m) {
        const int token = tt0 * 16 + m;
        const int k     = idxAll[m];
        const float4 qv = *(const float4*)(cbf + ((size_t)g * KK + k) * DC + lane * 4);
        const size_t off = ((size_t)token * GG + g) * DC + lane * 4;
        const float4 xv = *(const float4*)(xf + off);
        *(float4*)(out + off) = qv;
        float dx = xv.x - qv.x, dy = xv.y - qv.y, dz = xv.z - qv.z, dw = xv.w - qv.w;
        lsum += dx * dx + dy * dy + dz * dz + dw * dw;
    }
    #pragma unroll
    for (int m = 1; m < 32; m <<= 1) lsum += __shfl_xor(lsum, m, 32);
    if (lane == 0) atomicAdd(lossAcc, (double)lsum);
}

__global__ void vq_finalize(const double* __restrict__ lossAcc, float* __restrict__ out) {
    float mean = (float)(*lossAcc / (double)NELEM);
    out[NELEM]     = mean;   // commitment_loss
    out[NELEM + 1] = mean;   // codebook_loss (numerically identical)
}

extern "C" void kernel_launch(void* const* d_in, const int* in_sizes, int n_in,
                              void* d_out, int out_size, void* d_ws, size_t ws_size,
                              hipStream_t stream) {
    const float* x  = (const float*)d_in[0];   // (8,2048,1024) fp32
    const float* cb = (const float*)d_in[1];   // (8,1024,128)  fp32
    float* out = (float*)d_out;

    unsigned short* cbb = (unsigned short*)d_ws;          // 1M bf16
    float*          c2  = (float*)(cbb + GG * KK * DC);   // 8192 f32
    double*         acc = (double*)(c2 + GG * KK);        // 8B-aligned

    vq_prep_cb<<<(GG * KK) / 8, 256, 0, stream>>>(cb, cbb, c2, acc);

    const size_t smem = (size_t)KK * LROW * 2 + (size_t)KK * 4;   // 282,624 B < 320KB
    vq_main<<<512, 256, smem, stream>>>(cbb, c2, x, cb, out, acc);

    vq_finalize<<<1, 1, 0, stream>>>(acc, out);
}
